// BasicQuadRGBModel_26233660244075
// MI455X (gfx1250) — matrix operand order repaired and checked by reading, not compiled, not run on hardware
//
#include <hip/hip_runtime.h>

typedef __attribute__((ext_vector_type(16))) _Float16     v16h;
typedef __attribute__((ext_vector_type(8)))  float        v8f;
typedef __attribute__((ext_vector_type(8)))  unsigned int v8u;

static __device__ __forceinline__ unsigned short f2h(float f) {
  _Float16 h = (_Float16)f;
  return __builtin_bit_cast(unsigned short, h);
}
static __device__ __forceinline__ float h2f(unsigned short u) {
  _Float16 h = __builtin_bit_cast(_Float16, u);
  return (float)h;
}

// Implicit-GEMM 3x3 conv over an LDS tile region. One 16x16 WMMA N-tile
// (Cout <= 16) per 16-pixel M-tile; K is tap-major (k = tap*CIN + c, CIN even)
// so each A/B fragment dword is a single aligned 32-bit LDS read.
//   A (16x32 f16): lane m=l&15, half=l>>4; v<4 -> K=half*8+2v, v>=4 -> K=16+half*8+2(v-4)
//   B (32x16):     n=l&15; same K order as A
//   D (16x16 f32): lane col n=l&15, rows m=half*8+r
// No exec-divergence and no multi-object pointer selects anywhere in the hot
// loop: B fragments are hoisted, A K-padding reads a zeroed LDS word via a
// cndmask'd offset, and OOB stores select a dump *index* inside the SAME
// output array (32 pad slots appended), so every store is a plain ds_store_b16.
template<int CIN, int SOUT, int POUT, int KTOT>
static __device__ __forceinline__ void conv_layer(
    const unsigned short* featIn, int Sin,
    const unsigned short* wb,
    unsigned short* outBuf, bool doRelu, int outCh,
    const unsigned* zeroWord, int waveId, int lane)
{
  constexpr int NK   = (KTOT + 31) / 32;
  constexpr int KPAD = NK * 32;
  constexpr int NT   = (POUT + 15) / 16;
  const int m    = lane & 15;
  const int half = lane >> 4;

  // ---- B fragments: tile-invariant, weight image is zero-padded ----
  v16h Bf[NK];
  #pragma unroll
  for (int ks = 0; ks < NK; ++ks) {
    v8u bu;
    #pragma unroll
    for (int v = 0; v < 8; ++v)
      bu[v] = *(const unsigned*)&wb[m * KPAD + ks * 32 + ((v < 4) ? (half * 8 + v * 2) : (16 + half * 8 + (v - 4) * 2))];
    Bf[ks] = __builtin_bit_cast(v16h, bu);
  }

  // ---- per-lane A gather offsets (elements, rel. to pixel base) ----
  int  aOfs[NK * 8];
  bool aOk[NK * 8];
  #pragma unroll
  for (int ks = 0; ks < NK; ++ks) {
    #pragma unroll
    for (int v = 0; v < 8; ++v) {
      int k0  = ks * 32 + ((v < 4) ? (half * 8 + v * 2)
                                   : (16 + half * 8 + (v - 4) * 2));
      int tap = k0 / CIN, c = k0 - tap * CIN;          // c even
      aOfs[ks * 8 + v] = ((tap / 3) * Sin + (tap % 3)) * CIN + c;
      aOk[ks * 8 + v]  = (k0 < KTOT);
    }
  }

  const bool laneOk  = (m < outCh);
  const int  dumpIdx = POUT * outCh + lane;   // pad slots appended to outBuf
  for (int t = waveId; t < NT; t += 8) {
    int p  = t * 16 + m;
    int pc = (p < POUT) ? p : (POUT - 1);   // clamp addr; stores redirected
    int py = pc / SOUT, px = pc - py * SOUT;
    int base = (py * Sin + px) * CIN;
    v8f acc = {};
    #pragma unroll
    for (int ks = 0; ks < NK; ++ks) {
      v8u au;
      #pragma unroll
      for (int v = 0; v < 8; ++v) {
        const unsigned* ap = aOk[ks * 8 + v]
            ? (const unsigned*)(featIn + base + aOfs[ks * 8 + v])
            : zeroWord;
        au[v] = *ap;
      }
      v16h A = __builtin_bit_cast(v16h, au);
      acc = __builtin_amdgcn_wmma_f32_16x16x32_f16(
                false, A, false, Bf[ks], (short)0, acc, false, false);
    }
    #pragma unroll
    for (int r = 0; r < 8; ++r) {
      int  pr  = t * 16 + half * 8 + r;
      int  idx = (laneOk && pr < POUT) ? (pr * outCh + m) : dumpIdx;
      float val = acc[r];
      if (doRelu) val = fmaxf(val, 0.0f);
      outBuf[idx] = f2h(val);
    }
  }
}

__global__ __launch_bounds__(256)
void demosaic_fused(const float* __restrict__ mosaic,
                    const float* __restrict__ fw0, const float* __restrict__ fw1,
                    const float* __restrict__ fw2, const float* __restrict__ ww0,
                    const float* __restrict__ ww1, const float* __restrict__ ww2,
                    const float* __restrict__ cw0, float* __restrict__ out)
{
  // Halo pyramid: mosaic 24^2 -> conv1 22^2 -> conv2 20^2 -> conv3/green 18^2
  // -> chroma/output 16^2. Total static LDS ~78KB (<< 320KB/WGP).
  // Each conv output buffer has 32 extra "dump" slots for OOB store redirect.
  __shared__ float          sMosaic[576 * 4];         // f32 [pix][c]
  __shared__ unsigned short sMosaicH[576 * 4];        // f16 copy for WMMA A
  __shared__ unsigned short sF1[484 * 12 + 32];       // conv1 out (f then w chain)
  __shared__ unsigned short sF2[400 * 12 + 32];       // conv2 out
  __shared__ unsigned short sInterp[324 * 12 + 32];   // filter-chain conv3 out
  __shared__ unsigned short sW3[324 * 12 + 32];       // weight-chain conv3 out
  __shared__ float          sGreen[324 * 2];          // g0, g1
  __shared__ unsigned short sDiff[324 * 2];           // rb - green_rb (f16)
  __shared__ unsigned short sChroma[256 * 6 + 32];    // chroma conv out
  __shared__ unsigned short sWB0[2][16 * 64];         // fw0, ww0 (K=36 -> 64)
  __shared__ unsigned short sWB1[4][16 * 128];        // fw1, fw2, ww1, ww2 (K=108 -> 128)
  __shared__ unsigned short sWBc[16 * 32];            // cw0 (K=18 -> 32)
  __shared__ unsigned       sZero;                    // zero word for K-padding reads

  const int tid    = threadIdx.x;
  const int waveId = tid >> 5, lane = tid & 31;
  const int bx = blockIdx.x & 31;
  const int by = (blockIdx.x >> 5) & 31;
  const int b  = blockIdx.x >> 10;

  if (tid == 0) sZero = 0u;

  // ---- stage weights into WMMA-B layout: [n][k], k = tap*CIN + c, f16 ----
  for (int i = tid; i < 2048; i += 256) {
    int ws = i >> 10, e = i & 1023, n = e >> 6, k = e & 63;
    float v = 0.f;
    if (n < 12 && k < 36) {
      int tap = k >> 2, c = k & 3;
      v = (ws ? ww0 : fw0)[(n * 4 + c) * 9 + tap];
    }
    sWB0[ws][n * 64 + k] = f2h(v);
  }
  for (int i = tid; i < 8192; i += 256) {
    int ws = i >> 11, e = i & 2047, n = e >> 7, k = e & 127;
    float v = 0.f;
    if (n < 12 && k < 108) {
      int tap = k / 12, c = k - tap * 12;
      const float* src = (ws == 0) ? fw1 : (ws == 1) ? fw2 : (ws == 2) ? ww1 : ww2;
      v = src[(n * 12 + c) * 9 + tap];
    }
    sWB1[ws][n * 128 + k] = f2h(v);
  }
  for (int i = tid; i < 512; i += 256) {
    int n = i >> 5, k = i & 31;
    float v = 0.f;
    if (n < 6 && k < 18) {
      int tap = k >> 1, c = k & 1;
      v = cw0[(n * 2 + c) * 9 + tap];
    }
    sWBc[n * 32 + k] = f2h(v);
  }

  // ---- mosaic tile 24x24 (4-halo); zeroing OOB gives SAME padding free ----
  const int gy0 = by * 16 - 4, gx0 = bx * 16 - 4;
  for (int i = tid; i < 2304; i += 256) {
    int c = i / 576, pix = i - c * 576;
    int ly = pix / 24, lx = pix - ly * 24;
    int gy = gy0 + ly, gx = gx0 + lx;
    float v = 0.f;
    if (gy >= 0 && gy < 512 && gx >= 0 && gx < 512)
      v = mosaic[(((size_t)b * 4 + c) * 512 + gy) * 512 + gx];
    sMosaic[pix * 4 + c]  = v;
    sMosaicH[pix * 4 + c] = f2h(v);
  }
  __syncthreads();

  // filter chain: 3x (conv3x3 + relu)
  conv_layer<4, 22, 484, 36>(sMosaicH, 24, sWB0[0], sF1, true, 12, &sZero, waveId, lane);
  __syncthreads();
  conv_layer<12, 20, 400, 108>(sF1, 22, sWB1[0], sF2, true, 12, &sZero, waveId, lane);
  __syncthreads();
  conv_layer<12, 18, 324, 108>(sF2, 20, sWB1[1], sInterp, true, 12, &sZero, waveId, lane);
  __syncthreads();
  // weight chain: 3x (conv3x3 + relu)
  conv_layer<4, 22, 484, 36>(sMosaicH, 24, sWB0[1], sF1, true, 12, &sZero, waveId, lane);
  __syncthreads();
  conv_layer<12, 20, 400, 108>(sF1, 22, sWB1[2], sF2, true, 12, &sZero, waveId, lane);
  __syncthreads();
  conv_layer<12, 18, 324, 108>(sF2, 20, sWB1[3], sW3, true, 12, &sZero, waveId, lane);
  __syncthreads();

  // softmax over 12 channels (f32 math), green reconstruction, chroma diff
  for (int p = tid; p < 324; p += 256) {
    int py = p / 18, px = p - py * 18;
    float wv[12], mx = -3.4e38f;
    #pragma unroll
    for (int c = 0; c < 12; ++c) { wv[c] = h2f(sW3[p * 12 + c]); mx = fmaxf(mx, wv[c]); }
    float e[12], s = 0.f;
    #pragma unroll
    for (int c = 0; c < 12; ++c) { e[c] = __expf(wv[c] - mx); s += e[c]; }
    float inv = 1.0f / s;
    float g0 = 0.f, g1 = 0.f;
    #pragma unroll
    for (int c = 0; c < 6; ++c)  g0 += e[c] * inv * h2f(sInterp[p * 12 + c]);
    #pragma unroll
    for (int c = 6; c < 12; ++c) g1 += e[c] * inv * h2f(sInterp[p * 12 + c]);
    sGreen[p * 2 + 0] = g0;
    sGreen[p * 2 + 1] = g1;
    int mp = (py + 3) * 24 + (px + 3);
    sDiff[p * 2 + 0] = f2h(sMosaic[mp * 4 + 1] - g0);
    sDiff[p * 2 + 1] = f2h(sMosaic[mp * 4 + 2] - g1);
  }
  __syncthreads();

  // chroma conv 2->6, no relu
  conv_layer<2, 16, 256, 18>(sDiff, 18, sWBc, sChroma, false, 6, &sZero, waveId, lane);
  __syncthreads();

  // pixel-shuffle assembly: each thread emits one 2x2 quad per RGB channel
  {
    int p = tid, py = p >> 4, px = p & 15;
    int mp = (py + 4) * 24 + (px + 4);
    int gp = (py + 1) * 18 + (px + 1);
    float m0 = sMosaic[mp * 4 + 0], m1 = sMosaic[mp * 4 + 1];
    float m2 = sMosaic[mp * 4 + 2], m3 = sMosaic[mp * 4 + 3];
    float g0 = sGreen[gp * 2 + 0],  g1 = sGreen[gp * 2 + 1];
    float cp0 = h2f(sChroma[p * 6 + 0]) + m0;   // green_add = {m0,g1,m3,m0,g0,m3}
    float cp1 = h2f(sChroma[p * 6 + 1]) + g1;
    float cp2 = h2f(sChroma[p * 6 + 2]) + m3;
    float cp3 = h2f(sChroma[p * 6 + 3]) + m0;
    float cp4 = h2f(sChroma[p * 6 + 4]) + g0;
    float cp5 = h2f(sChroma[p * 6 + 5]) + m3;
    size_t Y = (size_t)(by * 16 + py) * 2;
    size_t X = (size_t)(bx * 16 + px) * 2;
    float* outR = out + (((size_t)b * 3 + 0) * 1024 + Y) * 1024 + X;
    float* outG = out + (((size_t)b * 3 + 1) * 1024 + Y) * 1024 + X;
    float* outB = out + (((size_t)b * 3 + 2) * 1024 + Y) * 1024 + X;
    *(float2*)(outR)        = make_float2(cp0, m1);
    *(float2*)(outR + 1024) = make_float2(cp1, cp2);
    *(float2*)(outG)        = make_float2(m0, g0);
    *(float2*)(outG + 1024) = make_float2(g1, m3);
    *(float2*)(outB)        = make_float2(cp3, cp4);
    *(float2*)(outB + 1024) = make_float2(m2, cp5);
  }
}

extern "C" void kernel_launch(void* const* d_in, const int* in_sizes, int n_in,
                              void* d_out, int out_size, void* d_ws, size_t ws_size,
                              hipStream_t stream) {
  const float* mosaic = (const float*)d_in[0];
  const float* fw0 = (const float*)d_in[1];
  const float* fw1 = (const float*)d_in[2];
  const float* fw2 = (const float*)d_in[3];
  const float* ww0 = (const float*)d_in[4];
  const float* ww1 = (const float*)d_in[5];
  const float* ww2 = (const float*)d_in[6];
  const float* cw0 = (const float*)d_in[7];
  // 16 images x (32x32) tiles of 16x16 half-res pixels each
  dim3 grid(16 * 32 * 32), block(256);
  demosaic_fused<<<grid, block, 0, stream>>>(mosaic, fw0, fw1, fw2,
                                             ww0, ww1, ww2, cw0,
                                             (float*)d_out);
}
